// HierarchicalLocalAttentionND_23304492548292
// MI455X (gfx1250) — matrix operand order repaired and checked by reading, not compiled, not run on hardware
//
#include <hip/hip_runtime.h>
#include <math.h>

// ---------------- problem constants ----------------
#define BB    2
#define LL    2048
#define NTOK  (BB * LL)       // 4096
#define CC    1024
#define HH    8
#define DD    128             // CC / HH
#define SS    33
#define MPROJ 160             // 24 wave rows + 128 query rows (logical)
#define MPROJP 256            // padded to a full 128-wide block tile x2
#define MSE1  256
#define KSE2  256

// ---------------- WMMA vector types ----------------
typedef __attribute__((ext_vector_type(16))) __bf16 v16bf;
typedef __attribute__((ext_vector_type(8)))  __bf16 v8bf;
typedef __attribute__((ext_vector_type(8)))  float  v8f;

union V16U { v16bf v; v8bf h[2]; };

// fp32 -> bf16 round-to-nearest-even (as raw u16)
__device__ __forceinline__ unsigned short f2bf(float f) {
  unsigned int u = __float_as_uint(f);
  unsigned int r = u + 0x7FFFu + ((u >> 16) & 1u);
  return (unsigned short)(r >> 16);
}

__device__ __forceinline__ float sigmoidf_(float v) { return 1.0f / (1.0f + __expf(-v)); }

// ---- CDNA5 async copy: global -> LDS, tracked by ASYNCcnt (CDNA5 ISA) ----
__device__ __forceinline__ void async_copy_b128(unsigned lds_addr, const void* gaddr) {
  asm volatile("global_load_async_to_lds_b128 %0, %1, off"
               :: "v"(lds_addr), "v"(gaddr) : "memory");
}
__device__ __forceinline__ void wait_async_le1() {
  asm volatile("s_wait_asynccnt 0x1" ::: "memory");
}
__device__ __forceinline__ void wait_async_le0() {
  asm volatile("s_wait_asynccnt 0x0" ::: "memory");
}

// ---------------- packing / conversion kernels ----------------
__global__ void conv_bf16_kernel(const float* __restrict__ s, unsigned short* __restrict__ d, int n) {
  int i = blockIdx.x * blockDim.x + threadIdx.x;
  if (i < n) d[i] = f2bf(s[i]);
}

// Wt[k*M + m] = bf16(W[m*K + k])   (W is M x K row-major)
__global__ void transpose_bf16_kernel(const float* __restrict__ W, unsigned short* __restrict__ Wt,
                                      int M, int K) {
  int i = blockIdx.x * blockDim.x + threadIdx.x;
  if (i >= M * K) return;
  int m = i / K, k = i - m * K;
  Wt[(size_t)k * M + m] = f2bf(W[i]);
}

// combined wave(24) + query(128) weights -> K-major bf16 (1024 x 256), rows 152..255 zero
__global__ void pack_wct_kernel(const float* __restrict__ w_wave, const float* __restrict__ w_query,
                                const float* __restrict__ b_wave, const float* __restrict__ b_query,
                                unsigned short* __restrict__ WCt, float* __restrict__ bc) {
  int i = blockIdx.x * blockDim.x + threadIdx.x;
  if (i < MPROJP)
    bc[i] = (i < 24) ? b_wave[i] : (i < 152 ? b_query[i - 24] : 0.0f);
  if (i >= MPROJP * CC) return;
  int j = i / CC, k = i - j * CC;
  float v = (j < 24) ? w_wave[j * CC + k] : (j < 152 ? w_query[(j - 24) * CC + k] : 0.0f);
  WCt[(size_t)k * MPROJP + j] = f2bf(v);
}

// A operand fragment from an LDS slab (32 tokens x 32 K halves), ISA 7.12.2 16-bit A layout:
// lanes 0-15 -> rows, K halves {0..7, 16..23}; lanes 16-31 -> rows, K halves {8..15, 24..31}
__device__ __forceinline__ v16bf lds_a_frag(const unsigned short* slabRow, int kb) {
  V16U u;
  u.h[0] = *(const v8bf*)(slabRow + kb);
  u.h[1] = *(const v8bf*)(slabRow + kb + 16);
  return u.v;
}

// ---------------- WMMA GEMM:  Y(NxM) = act( X(NxK) @ Wt(KxMpad) + bias ) ----------------
// Block = 256 threads (8 waves) computes a 32-token x 128-feature tile.
// Wave (wn, wm): wn in {0,1} 16-token half, wm in {0..3} 32-feature quarter (2 WMMA N-tiles).
// A slab (32x32 bf16) staged in LDS via double-buffered global_load_async_to_lds_b128.
// ACT: 0=none, 1=silu, 2=sigmoid, 3=mul[token,feat]*sigmoid(v)
template <int ACT>
__global__ void __launch_bounds__(256)
gemm_bf16_kernel(const __bf16* __restrict__ X,
                 const __bf16* __restrict__ Wt,
                 const float*  __restrict__ bias,
                 const float*  __restrict__ mul,
                 float*        __restrict__ Yf,
                 unsigned short* __restrict__ Ybf,
                 int M, int Mpad, int K) {
  __shared__ __align__(16) unsigned short Asmem[2][32 * 32];

  const int tid  = threadIdx.x;
  const int lane = tid & 31;
  const int wave = tid >> 5;
  const int wn = wave >> 2;                 // token half within block tile
  const int wm = wave & 3;                  // feature quarter within block tile
  const int ntBlock = blockIdx.y * 32;
  const int ft0 = blockIdx.x * 128 + wm * 32;
  const int ft1 = ft0 + 16;

  // LDS byte address of the staging buffers (generic addr low 32 bits = LDS offset)
  const unsigned lbase = (unsigned)(unsigned long long)(void*)&Asmem[0][0];
  // async-copy work assignment: threads 0..127 (waves 0..3, wave-uniform predicate)
  const int ar = tid >> 2, ac = tid & 3;
  const __bf16* grow = X + (size_t)(ntBlock + ar) * K + ac * 8;

  const int nk = K >> 5;
  if (tid < 128) async_copy_b128(lbase + ar * 64 + ac * 16, grow);   // k-step 0 -> buf 0

  const int kb = (lane & 16) ? 8 : 0;
  const unsigned short* slabRow0 = &Asmem[0][(wn * 16 + (lane & 15)) * 32];
  v8f acc0 = {}, acc1 = {};

  for (int i = 0; i < nk; ++i) {
    const int k0 = i << 5;
    if (i + 1 < nk) {
      if (tid < 128)
        async_copy_b128(lbase + ((unsigned)((i + 1) & 1)) * 2048u + ar * 64 + ac * 16,
                        grow + (k0 + 32));
      wait_async_le1();                       // current buffer's copies done
    } else {
      wait_async_le0();
    }
    __syncthreads();

    v16bf a = lds_a_frag(slabRow0 + (i & 1) * 1024, kb);
    const __bf16* wrow = Wt + (size_t)k0 * Mpad;
    v16bf b0, b1;
#pragma unroll
    for (int s = 0; s < 16; ++s) {
      const int kk = (s < 8) ? (kb + s) : (16 + kb + (s - 8));
      b0[s] = wrow[(size_t)kk * Mpad + ft0 + (lane & 15)];
      b1[s] = wrow[(size_t)kk * Mpad + ft1 + (lane & 15)];
    }
    if (i + 1 < nk) __builtin_prefetch(wrow + (size_t)32 * Mpad + ft0, 0, 1);

    acc0 = __builtin_amdgcn_wmma_f32_16x16x32_bf16(false, a, false, b0, (short)0, acc0, false, false);
    acc1 = __builtin_amdgcn_wmma_f32_16x16x32_bf16(false, a, false, b1, (short)0, acc1, false, false);

    __syncthreads();                          // buffer consumed by all waves
  }

  // epilogue (EXEC divergence only after all WMMAs)
  const int fA = ft0 + (lane & 15);
  const int fB = ft1 + (lane & 15);
  const int rowb = ntBlock + wn * 16 + ((lane & 16) ? 8 : 0);
  const float bsA = (bias && fA < M) ? bias[fA] : 0.0f;
  const float bsB = (bias && fB < M) ? bias[fB] : 0.0f;

  auto emit = [&](float v, int token, int feat) {
    size_t o = (size_t)token * M + feat;
    if (ACT == 1)      v = v * sigmoidf_(v);
    else if (ACT == 2) v = sigmoidf_(v);
    else if (ACT == 3) v = mul[o] * sigmoidf_(v);
    if (Yf)  Yf[o]  = v;
    if (Ybf) Ybf[o] = f2bf(v);
  };
#pragma unroll
  for (int r = 0; r < 8; ++r) {
    int token = rowb + r;
    if (fA < M) emit(acc0[r] + bsA, token, fA);
    if (fB < M) emit(acc1[r] + bsB, token, fB);
  }
}

// ---------------- wave parameters per token ----------------
__global__ void wave_params_kernel(const float* __restrict__ Y1,   // N x 160
                                   const float* __restrict__ wkey, // 16
                                   float* __restrict__ freq, float* __restrict__ decay,
                                   float* __restrict__ qk,
                                   float* __restrict__ favg, float* __restrict__ pavg) {
  int t = blockIdx.x * blockDim.x + threadIdx.x;
  if (t >= NTOK) return;
  const float* y = Y1 + (size_t)t * MPROJ;
  float fa = 0.0f, pa = 0.0f;
#pragma unroll
  for (int h = 0; h < HH; ++h) {
    float f  = sigmoidf_(y[h]) * 15.0f + 1.0f;
    float ph = tanhf(y[HH + h]) * 16.0f;
    float dc = sigmoidf_(y[2 * HH + h]) * 9.5f + 0.5f;
    freq[t * HH + h]  = f;
    decay[t * HH + h] = dc;
    fa += f; pa += ph;
    float q = 0.0f;
#pragma unroll
    for (int p = 0; p < 16; ++p) q += y[24 + h * 16 + p] * wkey[p];
    qk[t * HH + h] = q;
  }
  favg[t] = fa * 0.125f;
  pavg[t] = pa * 0.125f;
}

// ---------------- attention gather: one block per (head, token) ----------------
__global__ void attn_kernel(const float* __restrict__ x,      // B,L,C fp32
                            const float* __restrict__ freq, const float* __restrict__ decay,
                            const float* __restrict__ qk,
                            const float* __restrict__ favg, const float* __restrict__ pavg,
                            float* __restrict__ out0, unsigned short* __restrict__ out0b,
                            float* __restrict__ entpart) {
  int h = blockIdx.x;
  int token = blockIdx.y;
  int b = token / LL, l = token - b * LL;

  __shared__ float s_attn[SS];
  __shared__ int   s_idx[SS];

  if (threadIdx.x == 0) {
    const float NEG_INF = -__builtin_inff();
    float fa = favg[token], pa = pavg[token];
    float fr = freq[token * HH + h];
    float dc = decay[token * HH + h];
    float q  = qk[token * HH + h] * 0.25f;     // POS^{-1/2}
    float inv_dc = 1.0f / fmaxf(dc, 0.1f);

    float m = NEG_INF;
#pragma unroll
    for (int s = 0; s < SS; ++s) {
      float off = (float)(s - 16);
      float pos = (float)l + off * fa + pa;
      bool  v   = (pos >= 0.0f) && (pos < (float)LL);
      float lg  = v ? fabsf(off) * fr * q : NEG_INF;
      m = fmaxf(m, lg);
    }
    float den = 0.0f;
#pragma unroll
    for (int s = 0; s < SS; ++s) {
      float off = (float)(s - 16);
      float pos = (float)l + off * fa + pa;
      bool  v   = (pos >= 0.0f) && (pos < (float)LL);
      float lg  = v ? fabsf(off) * fr * q : NEG_INF;
      den += __expf(lg - m);
    }
    float inv_den = 1.0f / den;
    float asum = 0.0f;
#pragma unroll
    for (int s = 0; s < SS; ++s) {
      float off = (float)(s - 16);
      float pos = (float)l + off * fa + pa;
      bool  v   = (pos >= 0.0f) && (pos < (float)LL);
      int   i   = (int)pos;                    // trunc toward zero, like astype(int32)
      i = i < 0 ? 0 : (i > LL - 1 ? LL - 1 : i);
      float rel = fabsf(off) * fr;
      float lg  = v ? rel * q : NEG_INF;
      float a   = __expf(lg - m) * inv_den * __expf(-rel * inv_dc) * (v ? 1.0f : 0.0f);
      s_attn[s] = a;
      s_idx[s]  = i;
      asum += a;
    }
    float inv = 1.0f / (asum + 1e-8f);
    float ent = 0.0f;
#pragma unroll
    for (int s = 0; s < SS; ++s) {
      float at = s_attn[s] * inv;
      s_attn[s] = at;
      ent += at * __logf(at + 1e-8f);
    }
    entpart[token * HH + h] = ent;             // = -entropy contribution
  }
  __syncthreads();

  int d = threadIdx.x;                         // 0..127
  int c = h * DD + d;
  const float* xb = x + (size_t)b * LL * CC;
  float acc = 0.0f;
#pragma unroll
  for (int s = 0; s < SS; ++s)
    acc += s_attn[s] * xb[(size_t)s_idx[s] * CC + c];
  size_t o = (size_t)token * CC + c;
  out0[o]  = acc;
  out0b[o] = f2bf(acc);
}

// ---------------- deterministic entropy reduction ----------------
__global__ void ent_reduce_kernel(const float* __restrict__ part, float* __restrict__ out_scalar,
                                  int n, float scale) {
  __shared__ float sm[256];
  float a = 0.0f;
  for (int i = threadIdx.x; i < n; i += 256) a += part[i];   // fixed strided order
  sm[threadIdx.x] = a;
  __syncthreads();
  for (int s = 128; s > 0; s >>= 1) {
    if (threadIdx.x < s) sm[threadIdx.x] += sm[threadIdx.x + s];
    __syncthreads();
  }
  if (threadIdx.x == 0) out_scalar[0] = sm[0] * scale;
}

// ---------------- host orchestration ----------------
static inline size_t align_up(size_t v, size_t a) { return (v + a - 1) & ~(a - 1); }

extern "C" void kernel_launch(void* const* d_in, const int* in_sizes, int n_in,
                              void* d_out, int out_size, void* d_ws, size_t ws_size,
                              hipStream_t stream) {
  const float* x       = (const float*)d_in[0];
  const float* w_wave  = (const float*)d_in[1];
  const float* b_wave  = (const float*)d_in[2];
  const float* w_query = (const float*)d_in[3];
  const float* b_query = (const float*)d_in[4];
  const float* w_key   = (const float*)d_in[5];
  const float* w_out   = (const float*)d_in[6];
  const float* w_se1   = (const float*)d_in[7];
  const float* b_se1   = (const float*)d_in[8];
  const float* w_se2   = (const float*)d_in[9];
  const float* b_se2   = (const float*)d_in[10];
  float* outp = (float*)d_out;

  // workspace carve-up
  char* base = (char*)d_ws;
  size_t off = 0;
  auto take = [&](size_t bytes) { char* p = base + off; off = align_up(off + bytes, 256); return p; };
  unsigned short* Xb     = (unsigned short*)take((size_t)NTOK * CC * 2);
  unsigned short* WCt    = (unsigned short*)take((size_t)CC * MPROJP * 2);
  float*          bc     = (float*)         take((size_t)MPROJP * 4);
  unsigned short* Wse1t  = (unsigned short*)take((size_t)CC * MSE1 * 2);
  unsigned short* Wse2t  = (unsigned short*)take((size_t)KSE2 * CC * 2);
  unsigned short* Woutt  = (unsigned short*)take((size_t)CC * CC * 2);
  float*          Y1     = (float*)         take((size_t)NTOK * MPROJ * 4);
  float*          freq   = (float*)         take((size_t)NTOK * HH * 4);
  float*          decay  = (float*)         take((size_t)NTOK * HH * 4);
  float*          qk     = (float*)         take((size_t)NTOK * HH * 4);
  float*          favg   = (float*)         take((size_t)NTOK * 4);
  float*          pavg   = (float*)         take((size_t)NTOK * 4);
  float*          out0   = (float*)         take((size_t)NTOK * CC * 4);
  unsigned short* out0b  = (unsigned short*)take((size_t)NTOK * CC * 2);
  unsigned short* G1b    = (unsigned short*)take((size_t)NTOK * MSE1 * 2);
  unsigned short* hb     = (unsigned short*)take((size_t)NTOK * CC * 2);
  float*          entpt  = (float*)         take((size_t)NTOK * HH * 4);
  (void)ws_size; (void)in_sizes; (void)n_in; (void)out_size;

  // 1. conversions / packing
  {
    int n = NTOK * CC;
    conv_bf16_kernel<<<(n + 255) / 256, 256, 0, stream>>>(x, Xb, n);
  }
  pack_wct_kernel<<<(MPROJP * CC + 255) / 256, 256, 0, stream>>>(w_wave, w_query, b_wave, b_query, WCt, bc);
  transpose_bf16_kernel<<<(MSE1 * CC + 255) / 256, 256, 0, stream>>>(w_se1, Wse1t, MSE1, CC);
  transpose_bf16_kernel<<<(CC * KSE2 + 255) / 256, 256, 0, stream>>>(w_se2, Wse2t, CC, KSE2);
  transpose_bf16_kernel<<<(CC * CC + 255) / 256, 256, 0, stream>>>(w_out, Woutt, CC, CC);

  // 2. fused wave+query projection with SiLU -> Y1 (N x 160), weights padded to 256 cols
  gemm_bf16_kernel<1><<<dim3(MPROJP / 128, NTOK / 32), 256, 0, stream>>>(
      (const __bf16*)Xb, (const __bf16*)WCt, bc, nullptr, Y1, nullptr, MPROJ, MPROJP, CC);

  // 3. per-token wave parameters
  wave_params_kernel<<<(NTOK + 255) / 256, 256, 0, stream>>>(Y1, w_key, freq, decay, qk, favg, pavg);

  // 4. attention gather -> out0 (fp32 + bf16), entropy partials
  attn_kernel<<<dim3(HH, NTOK), DD, 0, stream>>>(x, freq, decay, qk, favg, pavg, out0, out0b, entpt);

  // 5. SE squeeze: silu(out0 @ w_se1.T + b_se1) -> bf16 only
  gemm_bf16_kernel<1><<<dim3(MSE1 / 128, NTOK / 32), 256, 0, stream>>>(
      (const __bf16*)out0b, (const __bf16*)Wse1t, b_se1, nullptr, nullptr, G1b, MSE1, MSE1, CC);

  // 6. SE excite fused with gating: out0 * sigmoid(G1 @ w_se2.T + b_se2) -> bf16
  gemm_bf16_kernel<3><<<dim3(CC / 128, NTOK / 32), 256, 0, stream>>>(
      (const __bf16*)G1b, (const __bf16*)Wse2t, b_se2, out0, nullptr, hb, CC, CC, KSE2);

  // 7. output projection -> d_out fp32 (N x C)
  gemm_bf16_kernel<0><<<dim3(CC / 128, NTOK / 32), 256, 0, stream>>>(
      (const __bf16*)hb, (const __bf16*)Woutt, nullptr, nullptr, outp, nullptr, CC, CC, CC);

  // 8. deterministic entropy scalar -> d_out[N*C]
  ent_reduce_kernel<<<1, 256, 0, stream>>>(entpt, outp + (size_t)NTOK * CC,
                                           NTOK * HH, 1.0f / (float)(NTOK * HH));
}
